// SDENet_62835371540888
// MI455X (gfx1250) — compile-verified, matrix operand order checked by reading, not run on hardware
//
#include <hip/hip_runtime.h>
#include <hip/hip_bf16.h>

// ---------------------------------------------------------------------------
// SDENet forward for MI455X (gfx1250, wave32, WMMA + async LDS copy).
// Diffusion GEMMs use v_wmma_f32_16x16x32_f16 (f16 A/B, f32 accum).
// Diffusion L3 stages weights in LDS via global_load_async_to_lds_b128.
// ---------------------------------------------------------------------------

typedef _Float16 v16h __attribute__((ext_vector_type(16)));
typedef _Float16 half8 __attribute__((ext_vector_type(8)));
typedef float    v8f  __attribute__((ext_vector_type(8)));

#define SDE_DT      0.05f
#define SDE_SQRTDT  0.22360680f
#define SDE_B       1024
#define SDE_S       2304

// ---------------- helpers ----------------

__device__ inline float wave_sum32(float v) {
#pragma unroll
  for (int o = 16; o > 0; o >>= 1) v += __shfl_xor(v, o, 32);
  return v;
}

// Load one WMMA f16 operand fragment (A row or B column) for 16x16x32.
// Layout (ISA 7.12.2): lanes 0-15 hold K=[0..7] in elems 0-7, K=[16..23] in
// elems 8-15; lanes 16-31 hold K=[8..15] and K=[24..31].
__device__ inline v16h sde_load_frag(const _Float16* __restrict__ row, int kc, int half) {
  const half8 lo = *(const half8*)(row + kc + 8 * half);
  const half8 hi = *(const half8*)(row + kc + 16 + 8 * half);
  v16h r;
#pragma unroll
  for (int i = 0; i < 8; ++i) { r[i] = lo[i]; r[8 + i] = hi[i]; }
  return r;
}

// ---------------- weight convert/transpose (fp32 KxN -> f16 NxK) ----------------

__global__ void sde_transpose_f16_kernel(const float* __restrict__ src, _Float16* __restrict__ dst,
                                         int K, int N) {
  int idx = blockIdx.x * blockDim.x + threadIdx.x;
  if (idx >= K * N) return;
  int k = idx / N, n = idx % N;
  dst[(size_t)n * K + k] = (_Float16)src[idx];
}

// ---------------- direct convolution (3x3, VALID) ----------------

__global__ void sde_conv3x3_kernel(const float* __restrict__ x, const float* __restrict__ w,
                                   const float* __restrict__ bias, float* __restrict__ y,
                                   int Cin, int Hin, int Win, int Cout, int Hout, int Wout,
                                   int stride) {
  int idx = blockIdx.x * blockDim.x + threadIdx.x;
  int total = SDE_B * Cout * Hout * Wout;
  if (idx >= total) return;
  int xw = idx % Wout; int t = idx / Wout;
  int yh = t % Hout;   t /= Hout;
  int co = t % Cout;   int b = t / Cout;
  float acc = bias[co];
  const float* wbase = w + (size_t)co * Cin * 9;
  const float* xbase = x + (size_t)b * Cin * Hin * Win + (yh * stride) * Win + (xw * stride);
  for (int ci = 0; ci < Cin; ++ci) {
    const float* xp = xbase + (size_t)ci * Hin * Win;
    const float* wp = wbase + ci * 9;
#pragma unroll
    for (int ky = 0; ky < 3; ++ky)
#pragma unroll
      for (int kx = 0; kx < 3; ++kx)
        acc += xp[ky * Win + kx] * wp[ky * 3 + kx];
  }
  y[idx] = acc;
}

// ---------------- GroupNorm + ReLU (one wave per (sample, group)) ----------------

__global__ void sde_gn_relu_kernel(float* __restrict__ data, const float* __restrict__ gamma,
                                   const float* __restrict__ beta, _Float16* __restrict__ out16,
                                   int C, int HW, int groups) {
  int wid  = (blockIdx.x * blockDim.x + threadIdx.x) >> 5;
  int lane = threadIdx.x & 31;
  int b = wid / groups, grp = wid % groups;
  int cpg = C / groups;
  int n = cpg * HW;
  float* base = data + ((size_t)b * C + (size_t)grp * cpg) * HW;
  float s = 0.f, ss = 0.f;
  for (int i = lane; i < n; i += 32) { float v = base[i]; s += v; ss += v * v; }
  s = wave_sum32(s); ss = wave_sum32(ss);
  float mean = s / (float)n;
  float var  = ss / (float)n - mean * mean;
  float inv  = rsqrtf(var + 1e-5f);
  for (int i = lane; i < n; i += 32) {
    int c = grp * cpg + i / HW;
    float v = (base[i] - mean) * inv * gamma[c] + beta[c];
    v = v > 0.f ? v : 0.f;
    base[i] = v;
    if (out16) out16[(size_t)b * C * HW + (size_t)grp * cpg * HW + i] = (_Float16)v;
  }
}

// ---------------- drift MLP: one wave per sample ----------------

__global__ void sde_drift_kernel(const float* __restrict__ state,
                                 const float* __restrict__ W1, const float* __restrict__ B1,
                                 const float* __restrict__ W2, const float* __restrict__ B2,
                                 const float* __restrict__ W3, const float* __restrict__ B3,
                                 const float* __restrict__ W4, const float* __restrict__ B4,
                                 const float* __restrict__ W5, const float* __restrict__ B5,
                                 const float* __restrict__ W6, const float* __restrict__ B6,
                                 float* __restrict__ drift) {
  int wid  = (blockIdx.x * blockDim.x + threadIdx.x) >> 5;
  int lane = threadIdx.x & 31;
  const float* srow = state + (size_t)wid * SDE_S;

  // layer 1: 2304 -> 8 (lanes partition K, then cross-lane reduce)
  float acc[8];
#pragma unroll
  for (int j = 0; j < 8; ++j) acc[j] = 0.f;
  for (int k = lane; k < SDE_S; k += 32) {
    float sv = srow[k];
    const float* wr = W1 + (size_t)k * 8;
#pragma unroll
    for (int j = 0; j < 8; ++j) acc[j] += sv * wr[j];
  }
  float h1[8];
#pragma unroll
  for (int j = 0; j < 8; ++j) {
    float v = wave_sum32(acc[j]) + B1[j];
    h1[j] = v > 0.f ? v : 0.f;
  }
  // layer 2: 8 -> 16 (redundant per lane; uniform weights -> scalar loads)
  float h2[16];
#pragma unroll
  for (int j = 0; j < 16; ++j) {
    float v = B2[j];
#pragma unroll
    for (int i = 0; i < 8; ++i) v += h1[i] * W2[i * 16 + j];
    h2[j] = v > 0.f ? v : 0.f;
  }
  // layer 3: 16 -> 32
  float h3[32];
#pragma unroll
  for (int j = 0; j < 32; ++j) {
    float v = B3[j];
#pragma unroll
    for (int i = 0; i < 16; ++i) v += h2[i] * W3[i * 32 + j];
    h3[j] = v > 0.f ? v : 0.f;
  }
  // layer 4: 32 -> 16
  float h4[16];
#pragma unroll
  for (int j = 0; j < 16; ++j) {
    float v = B4[j];
#pragma unroll
    for (int i = 0; i < 32; ++i) v += h3[i] * W4[i * 16 + j];
    h4[j] = v > 0.f ? v : 0.f;
  }
  // layer 5: 16 -> 8
  float h5[8];
#pragma unroll
  for (int j = 0; j < 8; ++j) {
    float v = B5[j];
#pragma unroll
    for (int i = 0; i < 16; ++i) v += h4[i] * W5[i * 8 + j];
    h5[j] = v > 0.f ? v : 0.f;
  }
  // layer 6: 8 -> 2304 (lanes partition outputs)
  float* drow = drift + (size_t)wid * SDE_S;
  for (int s = lane; s < SDE_S; s += 32) {
    float v = B6[s];
#pragma unroll
    for (int i = 0; i < 8; ++i) v += h5[i] * W6[(size_t)i * SDE_S + s];
    drow[s] = v > 0.f ? v : 0.f;
  }
}

// ---------------- generic WMMA GEMM + bias + ReLU (f16 in, f16 out) ----------------
// One 16x16 output tile per wave; A is MxK row-major, Wt is NxK row-major.
// Register-pipelined global b128 loads feeding v_wmma (codegen-verified).

__global__ void sde_gemm_relu_wmma_kernel(const _Float16* __restrict__ A,
                                          const _Float16* __restrict__ Wt,
                                          const float* __restrict__ bias,
                                          _Float16* __restrict__ out,
                                          int N, int K, int tilesN) {
  int wid  = (blockIdx.x * blockDim.x + threadIdx.x) >> 5;
  int lane = threadIdx.x & 31;
  int half = lane >> 4, l16 = lane & 15;
  int tm = wid / tilesN, tn = wid % tilesN;
  const _Float16* arow = A  + (size_t)(tm * 16 + l16) * K;
  const _Float16* brow = Wt + (size_t)(tn * 16 + l16) * K;
  v8f c = {};
  for (int kc = 0; kc < K; kc += 32) {
    __builtin_prefetch(arow + kc + 64, 0, 0);
    v16h af = sde_load_frag(arow, kc, half);
    v16h bf = sde_load_frag(brow, kc, half);
    c = __builtin_amdgcn_wmma_f32_16x16x32_f16(false, af, false, bf, (short)0, c, false, false);
  }
  int col = tn * 16 + l16;
  float bv = bias[col];
#pragma unroll
  for (int r = 0; r < 8; ++r) {
    int m = tm * 16 + half * 8 + r;
    float v = c[r] + bv;
    v = v > 0.f ? v : 0.f;
    out[(size_t)m * N + col] = (_Float16)v;
  }
}

// ---------------- diffusion L3 WMMA + einsum(bsk,bk) + Euler update ----------------
// Block = 4 waves. The block async-copies a 64x64 f16 weight strip (8 KB) of
// W3^T into LDS once (ASYNCcnt path), then each wave computes a 16x64 output
// strip: A fragments (h2 row tile, K=64) loaded once from global and reused
// across 4 N-tiles whose B fragments come from LDS (ds_load_b128 -> WMMA).
// Each 16x16 tile covers 4 state columns x 4 brownian dims; a group-of-4
// shuffle reduction performs the einsum("bsk,bk") contraction in-register.

#define SDE_NSTRIPS 144   // 576 N-tiles / 4 per strip

__global__ void sde_diff3_update_kernel(const _Float16* __restrict__ h2,
                                        const _Float16* __restrict__ w3t,
                                        const float* __restrict__ bias3,
                                        const float* __restrict__ drift,
                                        const float* __restrict__ dWt,
                                        float* __restrict__ state,
                                        _Float16* __restrict__ state16) {
  const int K = 64;
  __shared__ _Float16 ldsB[64 * 64];   // 8 KB: 64 N-columns x K=64
  int strip = blockIdx.x % SDE_NSTRIPS;
  int tmgrp = blockIdx.x / SDE_NSTRIPS;
  int tid   = threadIdx.x;
  int wave  = tid >> 5;
  int lane  = tid & 31;
  int half  = lane >> 4, l16 = lane & 15;
  int tm    = tmgrp * 4 + wave;

  // --- async-stage the weight strip: each of 128 threads moves 64 bytes ---
  {
    const _Float16* gsrc = w3t + (size_t)strip * 64 * K + (size_t)tid * 32;
    unsigned int lds0 = (unsigned int)(uintptr_t)(ldsB + (size_t)tid * 32);
    unsigned long long ga = (unsigned long long)(uintptr_t)gsrc;
    asm volatile("global_load_async_to_lds_b128 %0, %1, off\n\t"
                 "global_load_async_to_lds_b128 %0, %1, off offset:16\n\t"
                 "global_load_async_to_lds_b128 %0, %1, off offset:32\n\t"
                 "global_load_async_to_lds_b128 %0, %1, off offset:48"
                 :: "v"(lds0), "v"(ga) : "memory");
    asm volatile("s_wait_asynccnt 0x0" ::: "memory");
  }
  __syncthreads();

  // --- A fragments for this wave's 16-row block, reused across 4 N-tiles ---
  const _Float16* arow = h2 + (size_t)(tm * 16 + l16) * K;
  v16h a0 = sde_load_frag(arow, 0, half);
  v16h a1 = sde_load_frag(arow, 32, half);

#pragma unroll
  for (int nt = 0; nt < 4; ++nt) {
    const _Float16* brow = (const _Float16*)(ldsB + (size_t)(nt * 16 + l16) * K);
    v16h b0 = sde_load_frag(brow, 0, half);
    v16h b1 = sde_load_frag(brow, 32, half);
    v8f c = {};
    c = __builtin_amdgcn_wmma_f32_16x16x32_f16(false, a0, false, b0, (short)0, c, false, false);
    c = __builtin_amdgcn_wmma_f32_16x16x32_f16(false, a1, false, b1, (short)0, c, false, false);

    int tn   = strip * 4 + nt;
    int nn   = tn * 16 + l16;        // column in 9216-wide diffusion output
    float bv = bias3[nn];
    int k    = nn & 3;               // brownian index
    int scol = tn * 4 + (l16 >> 2);  // state column covered by this lane group
#pragma unroll
    for (int r = 0; r < 8; ++r) {
      int m = tm * 16 + half * 8 + r;
      float v = c[r] + bv;
      v = v > 0.f ? v : 0.f;
      v *= dWt[m * 4 + k] * SDE_SQRTDT;
      // reduce across the 4 brownian lanes of this group
      v += __shfl_xor(v, 1, 32);
      v += __shfl_xor(v, 2, 32);
      if ((lane & 3) == 0) {
        size_t si = (size_t)m * SDE_S + scol;
        float nv = state[si] + SDE_DT * drift[si] + v;
        state[si] = nv;
        state16[si] = (_Float16)nv;
      }
    }
  }
}

// ---------------- classifier + softmax: one wave per sample ----------------

__global__ void sde_classifier_kernel(const float* __restrict__ state,
                                      const float* __restrict__ fccw,
                                      const float* __restrict__ fccb,
                                      float* __restrict__ out) {
  int wid  = (blockIdx.x * blockDim.x + threadIdx.x) >> 5;
  int lane = threadIdx.x & 31;
  const float* srow = state + (size_t)wid * SDE_S;
  float acc[10];
#pragma unroll
  for (int j = 0; j < 10; ++j) acc[j] = 0.f;
  for (int s = lane; s < SDE_S; s += 32) {
    float sv = srow[s];
    const float* wr = fccw + (size_t)s * 10;
#pragma unroll
    for (int j = 0; j < 10; ++j) acc[j] += sv * wr[j];
  }
  float lg[10];
  float mx = -3.4e38f;
#pragma unroll
  for (int j = 0; j < 10; ++j) {
    lg[j] = wave_sum32(acc[j]) + fccb[j];
    mx = lg[j] > mx ? lg[j] : mx;
  }
  float denom = 0.f;
#pragma unroll
  for (int j = 0; j < 10; ++j) { lg[j] = __expf(lg[j] - mx); denom += lg[j]; }
  if (lane < 10) out[(size_t)wid * 10 + lane] = lg[lane] / denom;
}

// ---------------------------------------------------------------------------

extern "C" void kernel_launch(void* const* d_in, const int* in_sizes, int n_in,
                              void* d_out, int out_size, void* d_ws, size_t ws_size,
                              hipStream_t stream) {
  (void)in_sizes; (void)n_in; (void)out_size; (void)ws_size;

  const float* x  = (const float*)d_in[0];   // (1024,3,32,32)
  const float* dW = (const float*)d_in[1];   // (20,1024,4)
  const float* conv1_w = (const float*)d_in[2];  const float* conv1_b = (const float*)d_in[3];
  const float* gn1_g   = (const float*)d_in[4];  const float* gn1_b   = (const float*)d_in[5];
  const float* conv2_w = (const float*)d_in[6];  const float* conv2_b = (const float*)d_in[7];
  const float* gn2_g   = (const float*)d_in[8];  const float* gn2_b   = (const float*)d_in[9];
  const float* conv3_w = (const float*)d_in[10]; const float* conv3_b = (const float*)d_in[11];
  const float* gn3_g   = (const float*)d_in[12]; const float* gn3_b   = (const float*)d_in[13];
  const float* dr_w[6]; const float* dr_b[6];
  for (int i = 0; i < 6; ++i) { dr_w[i] = (const float*)d_in[14 + i]; dr_b[i] = (const float*)d_in[20 + i]; }
  const float* df_w1 = (const float*)d_in[26]; const float* df_w2 = (const float*)d_in[27];
  const float* df_w3 = (const float*)d_in[28];
  const float* df_b1 = (const float*)d_in[29]; const float* df_b2 = (const float*)d_in[30];
  const float* df_b3 = (const float*)d_in[31];
  const float* fcc_w = (const float*)d_in[32]; const float* fcc_b = (const float*)d_in[33];
  float* out = (float*)d_out;

  // scratch carve-out
  char* ws = (char*)d_ws;
  size_t off = 0;
  auto carve = [&](size_t bytes) -> char* {
    char* p = ws + off;
    off += (bytes + 255) & ~(size_t)255;
    return p;
  };
  float*    c1buf   = (float*)   carve((size_t)SDE_B * 16 * 30 * 30 * 4);
  float*    c2buf   = (float*)   carve((size_t)SDE_B * 32 * 14 * 14 * 4);
  float*    state   = (float*)   carve((size_t)SDE_B * SDE_S * 4);
  float*    driftb  = (float*)   carve((size_t)SDE_B * SDE_S * 4);
  _Float16* state16 = (_Float16*)carve((size_t)SDE_B * SDE_S * 2);
  _Float16* h1buf   = (_Float16*)carve((size_t)SDE_B * 128 * 2);
  _Float16* h2buf   = (_Float16*)carve((size_t)SDE_B * 64 * 2);
  _Float16* w1t     = (_Float16*)carve((size_t)128 * SDE_S * 2);
  _Float16* w2t     = (_Float16*)carve((size_t)64 * 128 * 2);
  _Float16* w3t     = (_Float16*)carve((size_t)9216 * 64 * 2);

  // --- convert diffusion weights to transposed f16 ---
  {
    int n1 = SDE_S * 128, n2 = 128 * 64, n3 = 64 * 9216;
    sde_transpose_f16_kernel<<<(n1 + 255) / 256, 256, 0, stream>>>(df_w1, w1t, SDE_S, 128);
    sde_transpose_f16_kernel<<<(n2 + 255) / 256, 256, 0, stream>>>(df_w2, w2t, 128, 64);
    sde_transpose_f16_kernel<<<(n3 + 255) / 256, 256, 0, stream>>>(df_w3, w3t, 64, 9216);
  }

  // --- feature extractor ---
  {
    int t1 = SDE_B * 16 * 30 * 30;
    sde_conv3x3_kernel<<<(t1 + 255) / 256, 256, 0, stream>>>(x, conv1_w, conv1_b, c1buf,
                                                             3, 32, 32, 16, 30, 30, 1);
    sde_gn_relu_kernel<<<(SDE_B * 8 * 32) / 128, 128, 0, stream>>>(c1buf, gn1_g, gn1_b,
                                                                   (_Float16*)nullptr, 16, 900, 8);
    int t2 = SDE_B * 32 * 14 * 14;
    sde_conv3x3_kernel<<<(t2 + 255) / 256, 256, 0, stream>>>(c1buf, conv2_w, conv2_b, c2buf,
                                                             16, 30, 30, 32, 14, 14, 2);
    sde_gn_relu_kernel<<<(SDE_B * 16 * 32) / 128, 128, 0, stream>>>(c2buf, gn2_g, gn2_b,
                                                                    (_Float16*)nullptr, 32, 196, 16);
    int t3 = SDE_B * 64 * 6 * 6;
    sde_conv3x3_kernel<<<(t3 + 255) / 256, 256, 0, stream>>>(c2buf, conv3_w, conv3_b, state,
                                                             32, 14, 14, 64, 6, 6, 2);
    // GN3 writes normalized state (fp32) in place and the f16 copy for WMMA
    sde_gn_relu_kernel<<<(SDE_B * 32 * 32) / 128, 128, 0, stream>>>(state, gn3_g, gn3_b,
                                                                    state16, 64, 36, 32);
  }

  // --- 20 Euler-Maruyama steps ---
  for (int t = 0; t < 20; ++t) {
    const float* dWt = dW + (size_t)t * SDE_B * 4;
    // drift MLP (VALU, wave per sample)
    sde_drift_kernel<<<(SDE_B * 32) / 128, 128, 0, stream>>>(
        state, dr_w[0], dr_b[0], dr_w[1], dr_b[1], dr_w[2], dr_b[2],
        dr_w[3], dr_b[3], dr_w[4], dr_b[4], dr_w[5], dr_b[5], driftb);
    // diffusion L1: (1024 x 2304) @ (2304 x 128)  -> 512 tiles, 1 per wave
    sde_gemm_relu_wmma_kernel<<<(512 * 32) / 128, 128, 0, stream>>>(
        state16, w1t, df_b1, h1buf, 128, SDE_S, 128 / 16);
    // diffusion L2: (1024 x 128) @ (128 x 64)     -> 256 tiles, 1 per wave
    sde_gemm_relu_wmma_kernel<<<(256 * 32) / 128, 128, 0, stream>>>(
        h1buf, w2t, df_b2, h2buf, 64, 128, 64 / 16);
    // diffusion L3 + dW contraction + state update:
    // 16 tm-groups x 144 N-strips, 4 waves/block, LDS-staged weights
    sde_diff3_update_kernel<<<16 * SDE_NSTRIPS, 128, 0, stream>>>(
        h2buf, w3t, df_b3, driftb, dWt, state, state16);
  }

  // --- classifier + softmax ---
  sde_classifier_kernel<<<(SDE_B * 32) / 128, 128, 0, stream>>>(state, fcc_w, fcc_b, out);
}
